// DeepseekV3TopkRouter_29506425323544
// MI455X (gfx1250) — matrix operand order repaired and checked by reading, not compiled, not run on hardware
//
#include <hip/hip_runtime.h>
#include <hip/hip_bf16.h>

#define HIDDEN 7168
#define NEXP   256
#define NGROUP 8
#define EPG    32      // experts per group
#define TOPKG  4
#define TOPK   8
#define MTILE  64      // tokens per workgroup
#define SCALING 2.5f

typedef __attribute__((ext_vector_type(16))) __bf16 v16bf;
typedef __attribute__((ext_vector_type(8)))  float  v8f;
typedef __attribute__((ext_vector_type(4)))  float  f32x4;

static __device__ __forceinline__ void pack8(v16bf& d, int base, f32x4 q0, f32x4 q1) {
    d[base + 0] = (__bf16)q0.x; d[base + 1] = (__bf16)q0.y;
    d[base + 2] = (__bf16)q0.z; d[base + 3] = (__bf16)q0.w;
    d[base + 4] = (__bf16)q1.x; d[base + 5] = (__bf16)q1.y;
    d[base + 6] = (__bf16)q1.z; d[base + 7] = (__bf16)q1.w;
}

// A fragment (16x32 bf16, wave32 layout): lane half h holds K = {h*8..h*8+7, 16+h*8..23+h*8}
// p points at row_base + k0 + h*8
static __device__ __forceinline__ v16bf load_a_frag(const float* __restrict__ p) {
    f32x4 q0 = *(const f32x4*)(p + 0);
    f32x4 q1 = *(const f32x4*)(p + 4);
    f32x4 q2 = *(const f32x4*)(p + 16);
    f32x4 q3 = *(const f32x4*)(p + 20);
    v16bf d; pack8(d, 0, q0, q1); pack8(d, 8, q2, q3); return d;
}

// B fragment (32x16 bf16): lane half h holds K = {h*8..h*8+7, 16+h*8..23+h*8}
// p points at expert_row_base + k0 + h*8
static __device__ __forceinline__ v16bf load_b_frag(const float* __restrict__ p) {
    f32x4 q0 = *(const f32x4*)(p + 0);
    f32x4 q1 = *(const f32x4*)(p + 4);
    f32x4 q2 = *(const f32x4*)(p + 16);
    f32x4 q3 = *(const f32x4*)(p + 20);
    v16bf d; pack8(d, 0, q0, q1); pack8(d, 8, q2, q3); return d;
}

__global__ __launch_bounds__(256, 1)
void deepseek_router_kernel(const float* __restrict__ x,
                            const float* __restrict__ wgt,
                            const float* __restrict__ bias,
                            int*   __restrict__ out_idx,
                            float* __restrict__ out_w,
                            int T) {
    __shared__ float sc[MTILE * NEXP];                 // 64 KB: logits -> scores_for_choice
    __shared__ float gsum[MTILE * NGROUP];
    __shared__ int   gsel[MTILE * TOPKG];
    __shared__ float candv[MTILE * TOPKG * TOPK];
    __shared__ int   candi[MTILE * TOPKG * TOPK];

    const int tid = threadIdx.x;
    const int w   = tid >> 5;        // wave 0..7 -> experts [w*32, w*32+32)
    const int l   = tid & 31;
    const int hh  = l >> 4;          // lane half
    const int mn  = l & 15;          // A row / B,C column within tile
    const int tb  = blockIdx.x * MTILE;

    // ---------------- GEMM phase: logits[MTILE][256] via bf16 WMMA ----------------
    const float* xp[4];
#pragma unroll
    for (int mt = 0; mt < 4; ++mt) {
        int row = tb + mt * 16 + mn;
        if (row > T - 1) row = T - 1;
        xp[mt] = x + (size_t)row * HIDDEN;
    }
    const float* bp0 = wgt + (size_t)(w * 32 + mn) * HIDDEN;
    const float* bp1 = bp0 + (size_t)16 * HIDDEN;

    v8f acc[4][2];
#pragma unroll
    for (int mt = 0; mt < 4; ++mt) { acc[mt][0] = {}; acc[mt][1] = {}; }

    for (int k0 = 0; k0 < HIDDEN; k0 += 32) {
        v16bf b0 = load_b_frag(bp0 + k0 + hh * 8);
        v16bf b1 = load_b_frag(bp1 + k0 + hh * 8);
#pragma unroll
        for (int mt = 0; mt < 4; ++mt) {
            v16bf a = load_a_frag(xp[mt] + k0 + hh * 8);
            acc[mt][0] = __builtin_amdgcn_wmma_f32_16x16x32_bf16(
                false, a, false, b0, (short)0, acc[mt][0], false, false);
            acc[mt][1] = __builtin_amdgcn_wmma_f32_16x16x32_bf16(
                false, a, false, b1, (short)0, acc[mt][1], false, false);
        }
    }

    // C layout: VGPR r -> M = r + 8*laneHalf ; N = lane&15. Store logits to LDS.
#pragma unroll
    for (int mt = 0; mt < 4; ++mt)
#pragma unroll
        for (int nt = 0; nt < 2; ++nt)
#pragma unroll
            for (int r = 0; r < 8; ++r)
                sc[(mt * 16 + hh * 8 + r) * NEXP + (w * 32 + nt * 16 + mn)] = acc[mt][nt][r];
    __syncthreads();

    // ---------------- scores_for_choice = sigmoid(logit) + bias, in place ----------------
    for (int i = tid; i < MTILE * NEXP; i += 256) {
        int e = i & (NEXP - 1);
        float s = 1.0f / (1.0f + __expf(-sc[i]));
        sc[i] = s + bias[e];
    }
    __syncthreads();

    // ---------------- group score = sum of top-2 within each group of 32 ----------------
    for (int i = tid; i < MTILE * NGROUP; i += 256) {
        int t = i >> 3, g = i & 7;
        const float* p = &sc[t * NEXP + g * EPG];
        float m1 = -1e30f, m2 = -1e30f;
        for (int j = 0; j < EPG; ++j) {
            float v = p[j];
            if (v > m1) { m2 = m1; m1 = v; }
            else if (v > m2) { m2 = v; }
        }
        gsum[i] = m1 + m2;
    }
    __syncthreads();

    // ---------------- top-4 groups per token ----------------
    if (tid < MTILE) {
        float v[NGROUP];
#pragma unroll
        for (int g = 0; g < NGROUP; ++g) v[g] = gsum[tid * NGROUP + g];
#pragma unroll
        for (int p = 0; p < TOPKG; ++p) {
            int best = 0; float bv = v[0];
#pragma unroll
            for (int g = 1; g < NGROUP; ++g)
                if (v[g] > bv) { bv = v[g]; best = g; }
            gsel[tid * TOPKG + p] = best;
            v[best] = -1e30f;
        }
    }
    __syncthreads();

    // ---------------- per selected group: local sorted top-8 (bubble insert) ----------------
    {
        int t = tid >> 2, p = tid & 3;                 // 64 tokens x 4 groups == 256 threads
        int g = gsel[t * TOPKG + p];
        const float* sp = &sc[t * NEXP + g * EPG];
        float v8[TOPK]; int i8[TOPK];
#pragma unroll
        for (int s = 0; s < TOPK; ++s) { v8[s] = -1e30f; i8[s] = -1; }
        for (int j = 0; j < EPG; ++j) {
            float v = sp[j]; int idx = g * EPG + j;
#pragma unroll
            for (int s = 0; s < TOPK; ++s) {
                if (v > v8[s]) {
                    float tv = v8[s]; v8[s] = v; v = tv;
                    int   ti = i8[s]; i8[s] = idx; idx = ti;
                }
            }
        }
#pragma unroll
        for (int s = 0; s < TOPK; ++s) {
            candv[(t * TOPKG + p) * TOPK + s] = v8[s];
            candi[(t * TOPKG + p) * TOPK + s] = i8[s];
        }
    }
    __syncthreads();

    // ---------------- merge 32 candidates -> top-8, normalize, scale, write ----------------
    if (tid < MTILE) {
        int t = tid;
        float v8[TOPK]; int i8[TOPK];
#pragma unroll
        for (int s = 0; s < TOPK; ++s) { v8[s] = -1e30f; i8[s] = -1; }
        for (int c = 0; c < TOPKG * TOPK; ++c) {
            float v = candv[t * TOPKG * TOPK + c];
            int idx  = candi[t * TOPKG * TOPK + c];
#pragma unroll
            for (int s = 0; s < TOPK; ++s) {
                if (v > v8[s]) {
                    float tv = v8[s]; v8[s] = v; v = tv;
                    int   ti = i8[s]; i8[s] = idx; idx = ti;
                }
            }
        }
        int tg = tb + t;
        if (tg < T) {
            float wv[TOPK]; float wsum = 0.0f;
#pragma unroll
            for (int s = 0; s < TOPK; ++s) {
                int e = i8[s];
                float sgm = sc[t * NEXP + e] - bias[e];   // recover sigmoid score
                wv[s] = sgm; wsum += sgm;
            }
            float inv = SCALING / (wsum + 1e-20f);
#pragma unroll
            for (int s = 0; s < TOPK; ++s) {
                out_idx[tg * TOPK + s] = i8[s];
                out_w[tg * TOPK + s]   = wv[s] * inv;
            }
        }
    }
}

extern "C" void kernel_launch(void* const* d_in, const int* in_sizes, int n_in,
                              void* d_out, int out_size, void* d_ws, size_t ws_size,
                              hipStream_t stream) {
    const float* x    = (const float*)d_in[0];
    const float* wgt  = (const float*)d_in[1];
    const float* bias = (const float*)d_in[2];
    const int T = in_sizes[0] / HIDDEN;

    // Tuple output: topk_indices [T,8] (int32) then topk_weights [T,8] (f32), flat-concatenated.
    int*   out_idx = (int*)d_out;
    float* out_w   = (float*)d_out + (size_t)T * TOPK;

    int blocks = (T + MTILE - 1) / MTILE;
    deepseek_router_kernel<<<blocks, 256, 0, stream>>>(x, wgt, bias, out_idx, out_w, T);
}
